// FunkNN_70119636074723
// MI455X (gfx1250) — matrix operand order repaired and checked by reading, not compiled, not run on hardware
//
#include <hip/hip_runtime.h>
#include <hip/hip_bf16.h>

// FunkNN: bicubic 9x9x3 crop + fused 10-layer MLP on the CDNA5 WMMA path.
// Weight-stream bound => M-tile = 32 points/block so each B fragment
// (global_load_b128 in hardware layout) feeds TWO v_wmma_f32_16x16x32_f16.
// Activations ping-pong in LDS (2 x 32x512 f16 = 64KB exactly).

typedef __attribute__((ext_vector_type(16))) _Float16 v16h;
typedef __attribute__((ext_vector_type(8)))  _Float16 v8h;
typedef __attribute__((ext_vector_type(8)))  float    v8f;

constexpr int NLAY = 10;
// Padded fan_in (K) and fan_out (N) per layer — all multiples of 32/16.
constexpr int KP[NLAY]   = {256,512,512,256,256,128,128, 64, 64, 64};
constexpr int NPAD[NLAY] = {512,512,256,256,128,128, 64, 64, 64, 16};
constexpr int KSRC[NLAY] = {243,512,512,256,256,128,128, 64, 64, 64};
constexpr int NSRC[NLAY] = {512,512,256,256,128,128, 64, 64, 64,  3};
// Element offsets of each layer's padded f16 weight block in the workspace.
constexpr int WOFF[NLAY] = {0,131072,393216,524288,589824,622592,638976,647168,651264,655360};
constexpr int WTOT = 656384;                 // f16 elements
constexpr int BOFF[NLAY] = {0,512,1024,1280,1536,1664,1792,1856,1920,1984};

constexpr int MT = 32;                       // points (rows) per block

// ---------------- weight / bias packing (f32 -> padded f16) ----------------

__global__ void pack_w_kernel(const float* __restrict__ src, _Float16* __restrict__ dst,
                              int Nsrc, int Ksrc, int Npad, int Kpad)
{
    int i = blockIdx.x * blockDim.x + threadIdx.x;
    if (i >= Npad * Kpad) return;
    int n = i / Kpad;
    int k = i - n * Kpad;
    float v = (n < Nsrc && k < Ksrc) ? src[n * Ksrc + k] : 0.0f;
    dst[i] = (_Float16)v;
}

__global__ void pack_b_kernel(const float* __restrict__ src, float* __restrict__ dst,
                              int Nsrc, int Npad)
{
    int i = blockIdx.x * blockDim.x + threadIdx.x;
    if (i >= Npad) return;
    dst[i] = (i < Nsrc) ? src[i] : 0.0f;
}

// ---------------- bicubic helpers (match reference semantics) ----------------

__device__ __forceinline__ float cubic_w(float s1)
{
    float s = s1 + 1e-6f;
    float a = fabsf(s);
    float w2 = ((-0.5f * a + 2.5f) * a - 4.0f) * a + 2.0f;
    float w1 = (1.5f * a - 2.5f) * a * a + 1.0f;
    float weq = (w1 + 3.0f * w2) * 0.25f;
    float o = (a < 2.0f) ? w2 : 0.0f;
    o = (a == 1.0f) ? weq : o;
    o = (a < 1.0f) ? w1 : o;
    return o;
}

__device__ __forceinline__ float reflectc(float x)
{
    x = (x > 127.5f) ? (255.0f - x) : x;   // 2*mx - x, mx = 127.5
    x = (x < -0.5f)  ? (-1.0f  - x) : x;   // 2*mn - x, mn = -0.5
    return x;
}

// Full-precision bicubic sample of one pixel / one channel (used for the
// mid-pixel residual recompute in the epilogue; identical math to phase 1).
__device__ float crop_pixel_chan(const float* __restrict__ image,
                                 const float* __restrict__ coord,
                                 int P, int px, int py, int c, float s1, float s2)
{
    const float half = 8.0f / 127.0f;
    int n = P >> 12;
    float dxg = 2.0f * coord[2 * P + 1];
    float dyg = 2.0f * coord[2 * P + 0];
    float bx = (px + 0.5f) * (2.0f / 9.0f) - 1.0f;
    float by = (py + 0.5f) * (2.0f / 9.0f) - 1.0f;
    float ix = (1.0f + (s1 * half * bx + dxg)) * 64.0f - 0.5f;
    float iy = (1.0f + (s2 * half * by + dyg)) * 64.0f - 0.5f;
    float ixb = floorf(ix), iyb = floorf(iy);
    const float* img = image + (size_t)n * 3 * 16384 + (size_t)c * 16384;
    float acc = 0.0f;
    #pragma unroll
    for (int i = 0; i < 4; ++i) {
        float tx = ixb + (float)(i - 1);
        float wxi = cubic_w(ix - tx);
        int rxi = (int)reflectc(tx);
        #pragma unroll
        for (int j = 0; j < 4; ++j) {
            float ty = iyb + (float)(j - 1);
            float wyj = cubic_w(iy - ty);
            int ryj = (int)reflectc(ty);
            acc += img[ryj * 128 + rxi] * (wxi * wyj);
        }
    }
    return acc;
}

// ---------------- WMMA fragment loaders ----------------
// A 16x32 f16 (ISA 7.12.2): lane l -> row l%16; K = 8*(l/16)+[0..7] and +16..23.
__device__ __forceinline__ v16h load_a_frag(const _Float16* buf, int K, int ks, int lane)
{
    int row = lane & 15;
    int kb  = ((lane >> 4) << 3) + (ks << 5);
    const _Float16* p = buf + row * K + kb;
    v8h lo = *(const v8h*)(p);
    v8h hi = *(const v8h*)(p + 16);
    return __builtin_shufflevector(lo, hi, 0,1,2,3,4,5,6,7,8,9,10,11,12,13,14,15);
}

// B 32x16 f16: lane l -> col N = 16*nt + l%16, K = 32*ks + 8*(l/16)+[0..7]
// and +16..23 (same order as A). Row-major [N][K] weights give two 16B loads.
__device__ __forceinline__ v16h load_b_frag(const _Float16* w, int K, int nt, int ks, int lane)
{
    int n  = (nt << 4) + (lane & 15);
    int kb = (ks << 5) + ((lane >> 4) << 3);
    const _Float16* p = w + n * K + kb;
    v8h lo = *(const v8h*)(p);
    v8h hi = *(const v8h*)(p + 16);
    return __builtin_shufflevector(lo, hi, 0,1,2,3,4,5,6,7,8,9,10,11,12,13,14,15);
}

// One MLP layer for a 32-row tile: D = relu(A*W^T + b). Each B fragment is
// loaded once and reused by the two 16-row A sub-tiles (2x arithmetic
// intensity on the weight stream).
__device__ __forceinline__ void mlp_layer(const _Float16* __restrict__ in,
                                          _Float16* __restrict__ outb,
                                          const _Float16* __restrict__ w,
                                          const float* __restrict__ bias,
                                          int K, int N, int wave, int lane)
{
    const int ksteps = K >> 5;
    const _Float16* in0 = in;               // rows 0..15
    const _Float16* in1 = in + 16 * K;      // rows 16..31
    for (int nt = wave; (nt << 4) < N; nt += 4) {        // wave-uniform: EXEC stays full
        v8f acc0 = {};
        v8f acc1 = {};
        for (int ks = 0; ks < ksteps; ks += 2) {         // K is always a multiple of 64
            v16h b0  = load_b_frag(w, K, nt, ks, lane);
            v16h a00 = load_a_frag(in0, K, ks, lane);
            v16h a10 = load_a_frag(in1, K, ks, lane);
            acc0 = __builtin_amdgcn_wmma_f32_16x16x32_f16(false, a00, false, b0,
                                                          (short)0, acc0, false, false);
            acc1 = __builtin_amdgcn_wmma_f32_16x16x32_f16(false, a10, false, b0,
                                                          (short)0, acc1, false, false);
            v16h b1  = load_b_frag(w, K, nt, ks + 1, lane);
            v16h a01 = load_a_frag(in0, K, ks + 1, lane);
            v16h a11 = load_a_frag(in1, K, ks + 1, lane);
            acc0 = __builtin_amdgcn_wmma_f32_16x16x32_f16(false, a01, false, b1,
                                                          (short)0, acc0, false, false);
            acc1 = __builtin_amdgcn_wmma_f32_16x16x32_f16(false, a11, false, b1,
                                                          (short)0, acc1, false, false);
        }
        const int col   = lane & 15;
        const int rbase = (lane >> 4) << 3;              // C/D layout: VGPR r -> row r(+8)
        const float bv  = bias[(nt << 4) + col];
        #pragma unroll
        for (int r = 0; r < 8; ++r) {
            float v0 = fmaxf(acc0[r] + bv, 0.0f);
            float v1 = fmaxf(acc1[r] + bv, 0.0f);
            outb[(rbase + r) * N + (nt << 4) + col]        = (_Float16)v0;
            outb[(16 + rbase + r) * N + (nt << 4) + col]   = (_Float16)v1;
        }
    }
}

// ---------------- fused crop + MLP kernel: one block = 32 points ----------------

__global__ __launch_bounds__(128)
void funknn_kernel(const float* __restrict__ coord,
                   const float* __restrict__ image,
                   const _Float16* __restrict__ wf16,
                   const float* __restrict__ bws,
                   const float* __restrict__ ws1,
                   const float* __restrict__ ws2,
                   float* __restrict__ out)
{
    __shared__ alignas(32) _Float16 bufA[MT * 512];   // 32 KB
    __shared__ alignas(32) _Float16 bufB[MT * 512];   // 32 KB  (total exactly 64 KB)

    const int tid  = threadIdx.x;
    const int lane = tid & 31;
    const int wave = tid >> 5;
    const int tile = blockIdx.x;

    const float s1 = ws1[0], s2 = ws2[0];
    const float half = 8.0f / 127.0f;

    // ---- Phase 1: bicubic crop of 32 patches (32 pts x 81 px) into bufA (f16) ----
    for (int t = tid; t < MT * 81; t += 128) {
        int p   = t / 81;
        int pix = t - p * 81;
        int py  = pix / 9;
        int px  = pix - py * 9;
        int P   = tile * MT + p;
        int n   = P >> 12;                       // batch index (bp = 4096)
        float dxg = 2.0f * coord[2 * P + 1];
        float dyg = 2.0f * coord[2 * P + 0];
        float bx = (px + 0.5f) * (2.0f / 9.0f) - 1.0f;
        float by = (py + 0.5f) * (2.0f / 9.0f) - 1.0f;
        float ix = (1.0f + (s1 * half * bx + dxg)) * 64.0f - 0.5f;
        float iy = (1.0f + (s2 * half * by + dyg)) * 64.0f - 0.5f;
        float ixb = floorf(ix), iyb = floorf(iy);

        float wx[4], wy[4]; int rx[4], ry[4];
        #pragma unroll
        for (int j = 0; j < 4; ++j) {
            float tx = ixb + (float)(j - 1);
            float ty = iyb + (float)(j - 1);
            wx[j] = cubic_w(ix - tx);
            wy[j] = cubic_w(iy - ty);
            rx[j] = (int)reflectc(tx);
            ry[j] = (int)reflectc(ty);
        }
        float a0 = 0.0f, a1 = 0.0f, a2 = 0.0f;
        const float* img = image + (size_t)n * 3 * 16384;  // L2-resident gathers
        #pragma unroll
        for (int i = 0; i < 4; ++i) {
            #pragma unroll
            for (int j = 0; j < 4; ++j) {
                int idx = ry[j] * 128 + rx[i];
                float wgt = wx[i] * wy[j];
                a0 += img[idx]         * wgt;
                a1 += img[idx + 16384] * wgt;
                a2 += img[idx + 32768] * wgt;
            }
        }
        bufA[p * 256 +       pix] = (_Float16)a0;   // layout h = [c*81 + y*9 + x]
        bufA[p * 256 +  81 + pix] = (_Float16)a1;
        bufA[p * 256 + 162 + pix] = (_Float16)a2;
    }
    // zero-pad K 243..255
    for (int t = tid; t < MT * 13; t += 128) {
        int p = t / 13;
        int k = 243 + (t - p * 13);
        bufA[p * 256 + k] = (_Float16)0.0f;
    }
    __syncthreads();

    // ---- Phase 2: layers 0..8 fused in LDS (ping-pong) ----
    const _Float16* cur = bufA;
    _Float16* nxt = bufB;
    #pragma unroll
    for (int l = 0; l < 9; ++l) {
        mlp_layer(cur, nxt, wf16 + WOFF[l], bws + BOFF[l], KP[l], NPAD[l], wave, lane);
        __syncthreads();
        const _Float16* t = cur; cur = nxt; nxt = const_cast<_Float16*>(t);
    }

    // ---- Phase 3: final 64 -> 3 (padded 16), + bias + mid-pixel residual ----
    if (wave == 0) {                              // wave-uniform: EXEC full for WMMA
        const _Float16* w9  = wf16 + WOFF[9];
        const _Float16* in0 = cur;                // rows 0..15
        const _Float16* in1 = cur + 16 * 64;      // rows 16..31
        v8f acc0 = {};
        v8f acc1 = {};
        #pragma unroll
        for (int ks = 0; ks < 2; ++ks) {          // K = 64
            v16h b  = load_b_frag(w9, 64, 0, ks, lane);
            v16h a0 = load_a_frag(in0, 64, ks, lane);
            v16h a1 = load_a_frag(in1, 64, ks, lane);
            acc0 = __builtin_amdgcn_wmma_f32_16x16x32_f16(false, a0, false, b,
                                                          (short)0, acc0, false, false);
            acc1 = __builtin_amdgcn_wmma_f32_16x16x32_f16(false, a1, false, b,
                                                          (short)0, acc1, false, false);
        }
        const int col   = lane & 15;
        const int rbase = (lane >> 4) << 3;
        if (col < 3) {                            // divergence only after WMMA
            const float bv = bws[BOFF[9] + col];
            #pragma unroll
            for (int r = 0; r < 8; ++r) {
                int row0 = rbase + r;
                int row1 = 16 + rbase + r;
                int P0 = tile * MT + row0;
                int P1 = tile * MT + row1;
                float mid0 = crop_pixel_chan(image, coord, P0, 4, 4, col, s1, s2);
                float mid1 = crop_pixel_chan(image, coord, P1, 4, 4, col, s1, s2);
                out[(size_t)P0 * 3 + col] = acc0[r] + bv + mid0;
                out[(size_t)P1 * 3 + col] = acc1[r] + bv + mid1;
            }
        }
    }
}

// ---------------- launch ----------------

extern "C" void kernel_launch(void* const* d_in, const int* in_sizes, int n_in,
                              void* d_out, int out_size, void* d_ws, size_t ws_size,
                              hipStream_t stream)
{
    (void)in_sizes; (void)n_in; (void)out_size; (void)ws_size;
    // setup_inputs order: coordinate, image, Ws[0..9], Bs[0..9], ws1, ws2
    const float* coord = (const float*)d_in[0];
    const float* image = (const float*)d_in[1];
    const float* ws1   = (const float*)d_in[22];
    const float* ws2   = (const float*)d_in[23];

    _Float16* wf16 = (_Float16*)d_ws;                                   // 1.28 MB
    float*    bws  = (float*)((char*)d_ws + (size_t)WTOT * sizeof(_Float16));

    for (int l = 0; l < NLAY; ++l) {
        int tot = NPAD[l] * KP[l];
        pack_w_kernel<<<(tot + 255) / 256, 256, 0, stream>>>(
            (const float*)d_in[2 + l], wf16 + WOFF[l], NSRC[l], KSRC[l], NPAD[l], KP[l]);
        pack_b_kernel<<<(NPAD[l] + 255) / 256, 256, 0, stream>>>(
            (const float*)d_in[12 + l], bws + BOFF[l], NSRC[l], NPAD[l]);
    }
    // 16384 points / 32 per tile = 512 blocks, 128 threads (4 waves) each.
    funknn_kernel<<<512, 128, 0, stream>>>(coord, image, wf16, bws, ws1, ws2, (float*)d_out);
}
